// Cross_MultiAttention_54322746360227
// MI455X (gfx1250) — matrix operand order repaired and checked by reading, not compiled, not run on hardware
//
#include <hip/hip_runtime.h>
#include <hip/hip_bf16.h>
#include <math.h>

typedef __attribute__((ext_vector_type(16))) _Float16 v16h;
typedef __attribute__((ext_vector_type(8)))  _Float16 v8h;
typedef __attribute__((ext_vector_type(8)))  float    v8f;

#define EMB   512
#define HEADS 8
#define DEPTH 64
#define SEQ   256
#define SCALEF 0.044194173824159216f   // 512^-0.5

// ws float layout
#define OFF_AQ 0        // [6][512]
#define OFF_CQ 3072     // [512]
#define OFF_BK 3584     // [12][512]
#define OFF_CK 9728     // [512]
#define OFF_BV 10240    // [12][512]
#define OFF_CV 16384    // [512]

// ---------------------------------------------------------------------------
// Kernel 1: fold linear chains:  Aq = W_emb @ Wq, cq = bq + b_emb @ Wq, etc.
// ---------------------------------------------------------------------------
__global__ void combine_weights(const float* __restrict__ W_emb,
                                const float* __restrict__ b_emb,
                                const float* __restrict__ W_emb2,
                                const float* __restrict__ b_emb2,
                                const float* __restrict__ Wq, const float* __restrict__ bq,
                                const float* __restrict__ Wk, const float* __restrict__ bk,
                                const float* __restrict__ Wv, const float* __restrict__ bv,
                                float* __restrict__ ws) {
  int j = blockIdx.x * blockDim.x + threadIdx.x;   // output column 0..511
  if (j >= EMB) return;
  // Aq[c][j], cq[j]
  for (int c = 0; c < 6; ++c) {
    float a = 0.f;
    for (int i = 0; i < EMB; ++i) a += W_emb[c * EMB + i] * Wq[i * EMB + j];
    ws[OFF_AQ + c * EMB + j] = a;
  }
  float cqv = bq[j];
  for (int i = 0; i < EMB; ++i) cqv += b_emb[i] * Wq[i * EMB + j];
  ws[OFF_CQ + j] = cqv;
  // Bk/ck, Bv/cv from W_emb2 (12 in-channels)
  for (int c = 0; c < 12; ++c) {
    float ak = 0.f, av = 0.f;
    for (int i = 0; i < EMB; ++i) {
      float w2 = W_emb2[c * EMB + i];
      ak += w2 * Wk[i * EMB + j];
      av += w2 * Wv[i * EMB + j];
    }
    ws[OFF_BK + c * EMB + j] = ak;
    ws[OFF_BV + c * EMB + j] = av;
  }
  float ckv = bk[j], cvv = bv[j];
  for (int i = 0; i < EMB; ++i) {
    ckv += b_emb2[i] * Wk[i * EMB + j];
    cvv += b_emb2[i] * Wv[i * EMB + j];
  }
  ws[OFF_CK + j] = ckv;
  ws[OFF_CV + j] = cvv;
}

// ---------------------------------------------------------------------------
// Kernel 2: init output with projection bias (atomics accumulate onto this)
// ---------------------------------------------------------------------------
__global__ void init_out(float* __restrict__ out, const float* __restrict__ bp) {
  int i = blockIdx.x * blockDim.x + threadIdx.x;
  if (i < 8 * 6 * 128 * 128) out[i] = bp[(i >> 14) % 6];   // 128*128 = 2^14
}

// ---------------------------------------------------------------------------
// Kernel 3: fused per-(block,head) cross attention (both Q1 and Q2 passes)
// grid = (512 blocks, 8 heads), 512 threads (16 wave32s, 1 row-tile each)
// ---------------------------------------------------------------------------
__global__ __launch_bounds__(512)
void cross_attn(const float* __restrict__ img1, const float* __restrict__ img2,
                const float* __restrict__ ws, const float* __restrict__ Wp,
                float* __restrict__ out) {
  __shared__ _Float16 bufA[SEQ * DEPTH];        // 32 KB: Ktt[j][d] then Vt[d][j]
  __shared__ float    fcat[SEQ * 12];           // 12 KB
  __shared__ _Float16 stage[16][16 * 32];       // 16 KB: per-wave A-frag staging

  const int blk  = blockIdx.x;                  // 0..511
  const int h    = blockIdx.y;                  // 0..7
  const int tid  = threadIdx.x;
  const int wave = tid >> 5;
  const int lane = tid & 31;
  const int m0   = wave * 16;                   // query row-tile base
  const int hi   = lane >> 4;                   // half-wave select
  const int ln   = lane & 15;

  const int bb = blk >> 6, ih = (blk >> 3) & 7, iw = blk & 7;

  const float* Aq = ws + OFF_AQ;
  const float* cq = ws + OFF_CQ;
  const float* Bk = ws + OFF_BK;
  const float* ck = ws + OFF_CK;
  const float* Bv = ws + OFF_BV;
  const float* cv = ws + OFF_CV;

  // ---- stage token features: fcat[j][c], c<6 from img1, c>=6 from img2 ----
  for (int e = tid; e < SEQ * 12; e += 512) {
    int j = e / 12, c = e % 12;
    int y = j >> 4, x = j & 15;
    const float* src = (c < 6) ? img1 : img2;
    int cc = (c < 6) ? c : c - 6;
    fcat[e] = src[(((size_t)bb * 6 + cc) * 128 + ih * 16 + y) * 128 + iw * 16 + x];
  }
  __syncthreads();

  for (int attn = 0; attn < 2; ++attn) {
    // ---- build K^T tile in LDS: Ktt[j][d]  (K = fcat @ Bk + ck) -----------
    if (attn) __syncthreads();                  // prior pass readers done
    for (int e = tid; e < SEQ * DEPTH; e += 512) {
      int j = e >> 6, d = e & 63;
      float acc = ck[h * DEPTH + d];
      #pragma unroll
      for (int c = 0; c < 12; ++c)
        acc += fcat[j * 12 + c] * Bk[c * EMB + h * DEPTH + d];
      bufA[e] = (_Float16)acc;
    }
    __syncthreads();

    // ---- Q A-fragments on the fly (scale folded in) -----------------------
    v16h qa[2];
    {
      int m = m0 + ln;                          // query row
      int base = hi * 8;                        // A-layout K half select
      #pragma unroll
      for (int kk = 0; kk < 2; ++kk) {
        #pragma unroll
        for (int e = 0; e < 16; ++e) {
          int klocal = 2 * base + e;
          int d = h * DEPTH + kk * 32 + klocal;
          float acc = cq[d];
          #pragma unroll
          for (int c = 0; c < 6; ++c)
            acc += fcat[m * 12 + 6 * attn + c] * Aq[c * EMB + d];
          qa[kk][e] = (_Float16)(acc * SCALEF);
        }
      }
    }

    // ---- S = Q K^T : 16 col-tiles, K-dim 64 = 2 chained WMMAs -------------
    v8f sc[16];
    #pragma unroll
    for (int jt = 0; jt < 16; ++jt) {
      const _Float16* pb = &bufA[(jt * 16 + ln) * DEPTH + hi * 16];
      v16h b0 = *(const v16h*)(pb);             // K 0..31 slice for this lane
      v16h b1 = *(const v16h*)(pb + 32);        // K 32..63
      v8f c0 = {};
      c0 = __builtin_amdgcn_wmma_f32_16x16x32_f16(false, qa[0], false, b0,
                                                  (short)0, c0, false, false);
      c0 = __builtin_amdgcn_wmma_f32_16x16x32_f16(false, qa[1], false, b1,
                                                  (short)0, c0, false, false);
      sc[jt] = c0;
    }
    __syncthreads();                            // all waves done reading Ktt

    // ---- softmax over full rows (regs + width-16 shuffles) ----------------
    #pragma unroll
    for (int r = 0; r < 8; ++r) {
      float m = -1e30f;
      #pragma unroll
      for (int jt = 0; jt < 16; ++jt) m = fmaxf(m, sc[jt][r]);
      for (int off = 1; off < 16; off <<= 1) m = fmaxf(m, __shfl_xor(m, off, 16));
      float s = 0.f;
      #pragma unroll
      for (int jt = 0; jt < 16; ++jt) {
        float v = __expf(sc[jt][r] - m);
        sc[jt][r] = v;
        s += v;
      }
      for (int off = 1; off < 16; off <<= 1) s += __shfl_xor(s, off, 16);
      float inv = 1.0f / s;
      #pragma unroll
      for (int jt = 0; jt < 16; ++jt) sc[jt][r] *= inv;
    }

    // ---- rebuild LDS buffer as V^T: Vt[d][j]  (V = fcat @ Bv + cv) --------
    for (int e = tid; e < SEQ * DEPTH; e += 512) {
      int d = e >> 8, j = e & 255;
      float acc = cv[h * DEPTH + d];
      #pragma unroll
      for (int c = 0; c < 12; ++c)
        acc += fcat[j * 12 + c] * Bv[c * EMB + h * DEPTH + d];
      bufA[d * SEQ + j] = (_Float16)acc;
    }
    __syncthreads();

    // ---- O = att @ V : 8 K-chunks of 32 keys, 4 depth col-tiles -----------
    v8f oacc[4] = {v8f{}, v8f{}, v8f{}, v8f{}};
    _Float16* st = &stage[wave][0];             // [16 rows][32 cols] f16
    for (int cc = 0; cc < 8; ++cc) {
      // C-layout f32 scores -> f16 A-fragment via per-wave LDS staging
      #pragma unroll
      for (int t = 0; t < 2; ++t) {
        int jt = 2 * cc + t;
        #pragma unroll
        for (int r = 0; r < 8; ++r)
          st[(r + 8 * hi) * 32 + t * 16 + ln] = (_Float16)sc[jt][r];
      }
      const _Float16* pa = &st[ln * 32 + hi * 16];
      v8h lo = *(const v8h*)(pa);               // K {0..7 | 16..23}
      v8h hh = *(const v8h*)(pa + 8);           // K {8..15 | 24..31}
      v16h af;
      #pragma unroll
      for (int e = 0; e < 8; ++e) { af[e] = lo[e]; af[8 + e] = hh[e]; }
      #pragma unroll
      for (int dt = 0; dt < 4; ++dt) {
        v16h bf = *(const v16h*)&bufA[(dt * 16 + ln) * SEQ + cc * 32 + hi * 16];
        oacc[dt] = __builtin_amdgcn_wmma_f32_16x16x32_f16(false, af, false, bf,
                                                          (short)0, oacc[dt],
                                                          false, false);
      }
    }

    // ---- fused output projection: proj[o,s] += sum_d O[s,d]*Wp[o,cbase+d] -
    const int cbase = attn * EMB + h * DEPTH;
    #pragma unroll
    for (int o = 0; o < 6; ++o) {
      float p[8];
      #pragma unroll
      for (int r = 0; r < 8; ++r) {
        float s = 0.f;
        #pragma unroll
        for (int dt = 0; dt < 4; ++dt)
          s += oacc[dt][r] * Wp[o * (2 * EMB) + cbase + dt * 16 + ln];
        for (int off = 1; off < 16; off <<= 1) s += __shfl_xor(s, off, 16);
        p[r] = s;
      }
      if (ln == 0) {                            // lane 0: rows r, lane 16: rows r+8
        #pragma unroll
        for (int r = 0; r < 8; ++r) {
          int srow = m0 + r + 8 * hi;
          int y = srow >> 4, x = srow & 15;
          atomicAdd(&out[(((size_t)bb * 6 + o) * 128 + ih * 16 + y) * 128 + iw * 16 + x],
                    p[r]);
        }
      }
    }
  }
}

// ---------------------------------------------------------------------------
extern "C" void kernel_launch(void* const* d_in, const int* in_sizes, int n_in,
                              void* d_out, int out_size, void* d_ws, size_t ws_size,
                              hipStream_t stream) {
  const float* img1   = (const float*)d_in[0];
  const float* img2   = (const float*)d_in[1];
  const float* W_emb  = (const float*)d_in[2];
  const float* b_emb  = (const float*)d_in[3];
  const float* W_emb2 = (const float*)d_in[4];
  const float* b_emb2 = (const float*)d_in[5];
  const float* Wq     = (const float*)d_in[6];
  const float* bq     = (const float*)d_in[7];
  const float* Wk     = (const float*)d_in[8];
  const float* bk     = (const float*)d_in[9];
  const float* Wv     = (const float*)d_in[10];
  const float* bv     = (const float*)d_in[11];
  const float* Wp     = (const float*)d_in[12];
  const float* bp     = (const float*)d_in[13];
  float* out = (float*)d_out;
  float* ws  = (float*)d_ws;

  combine_weights<<<2, 256, 0, stream>>>(W_emb, b_emb, W_emb2, b_emb2,
                                         Wq, bq, Wk, bk, Wv, bv, ws);
  init_out<<<(8 * 6 * 128 * 128 + 255) / 256, 256, 0, stream>>>(out, bp);
  dim3 grid(512, HEADS);
  cross_attn<<<grid, 512, 0, stream>>>(img1, img2, ws, Wp, out);
}
